// HeteroFraudGNN_8443905704157
// MI455X (gfx1250) — matrix operand order repaired and checked by reading, not compiled, not run on hardware
//
#include <hip/hip_runtime.h>
#include <hip/hip_bf16.h>

// ---------------- problem constants ----------------
#define NU   100000
#define NT   400000
#define NE   400000
#define HDIM 128
#define NLAY 3
#define OUTD 2
#define HID2 64
#define EPS  1e-5f
#define LDA  136   // padded LDS row stride in bf16 elements (breaks bank conflicts)

// ---------------- CDNA5 WMMA types ----------------
typedef __bf16 v16bf __attribute__((ext_vector_type(16)));
typedef __bf16 v2bf  __attribute__((ext_vector_type(2)));
typedef float  v8f   __attribute__((ext_vector_type(8)));
typedef unsigned int v4u __attribute__((ext_vector_type(4)));

union AFrag { v16bf v; v4u q[2]; };

// pack two f32 -> packed bf16 pair (native cvt)
static __device__ __forceinline__ unsigned int pack2bf(float x, float y) {
  v2bf t;
  t[0] = (__bf16)x;
  t[1] = (__bf16)y;
  return __builtin_bit_cast(unsigned int, t);
}
static __device__ __forceinline__ unsigned short bfbits(float f) {
  __bf16 b = (__bf16)f;
  return __builtin_bit_cast(unsigned short, b);
}
// pack float4 -> 4 bf16 and store 8B to LDS
static __device__ __forceinline__ void store4bf(unsigned short* p, float4 v) {
  unsigned int lo = pack2bf(v.x, v.y);
  unsigned int hi = pack2bf(v.z, v.w);
  unsigned long long q = ((unsigned long long)hi << 32) | (unsigned long long)lo;
  *reinterpret_cast<unsigned long long*>(p) = q;
}

// ---------------- kernels ----------------

// Wt[n*K + k] = bf16(W[k*N + n])  (pre-transpose weights into B-frag layout)
__global__ __launch_bounds__(256) void k_prep_w(const float* __restrict__ W,
                                                unsigned short* __restrict__ Wt,
                                                int K, int N) {
  int t = blockIdx.x * 256 + threadIdx.x;
  if (t < K * N) {
    int k = t / N, n = t % N;     // reads coalesced along n
    Wt[n * K + k] = bfbits(W[t]);
  }
}

// h[r, :] = emb[ids[r], :]   (one 32-lane wave per row, float4 per lane)
__global__ __launch_bounds__(256) void k_gather(const float* __restrict__ emb,
                                                const int* __restrict__ ids,
                                                float* __restrict__ h, int n) {
  int r = blockIdx.x * 8 + (threadIdx.x >> 5);
  int lane = threadIdx.x & 31;
  if (r < n) {
    int v = ids[r];
    float4 val = reinterpret_cast<const float4*>(emb + (size_t)v * HDIM)[lane];
    reinterpret_cast<float4*>(h + (size_t)r * HDIM)[lane] = val;
  }
}

// cnt[dst[e]] += 1
__global__ __launch_bounds__(256) void k_count(const int* __restrict__ dst,
                                               float* __restrict__ cnt, int e) {
  int t = blockIdx.x * 256 + threadIdx.x;
  if (t < e) atomicAdd(&cnt[dst[t]], 1.0f);
}

// agg[dst[e], :] += hsrc[src[e], :]   (one wave per edge)
__global__ __launch_bounds__(256) void k_scatter(const float* __restrict__ hsrc,
                                                 float* __restrict__ agg,
                                                 const int* __restrict__ src,
                                                 const int* __restrict__ dst, int e) {
  int edge = blockIdx.x * 8 + (threadIdx.x >> 5);
  int lane = threadIdx.x & 31;
  if (edge < e) {
    int s = src[edge], d = dst[edge];
    float4 v = reinterpret_cast<const float4*>(hsrc + (size_t)s * HDIM)[lane];
    float* dp = agg + (size_t)d * HDIM + lane * 4;
    atomicAdd(dp + 0, v.x);
    atomicAdd(dp + 1, v.y);
    atomicAdd(dp + 2, v.z);
    atomicAdd(dp + 3, v.w);
  }
}

// h[r,:] = relu(BN( (agg[r,:]/max(cnt,1)) @ Wl + bl + h[r,:] @ Wr ))  -- in place
// 128 rows per block, 8 waves, wave w owns output columns [16w, 16w+16).
// WlT/WrT are pre-transposed bf16 [n][k] so B-frags are two b128 loads each.
__global__ __launch_bounds__(256) void k_gemm_bn(
    int n, float* __restrict__ h, const float* __restrict__ agg,
    const float* __restrict__ cnt,
    const unsigned short* __restrict__ WlT, const float* __restrict__ bl,
    const unsigned short* __restrict__ WrT,
    const float* __restrict__ gma, const float* __restrict__ bta,
    const float* __restrict__ mu, const float* __restrict__ var) {
  __shared__ unsigned short sh[128 * LDA];  // h rows, bf16
  __shared__ unsigned short sa[128 * LDA];  // normalized agg rows, bf16
  const int tid = threadIdx.x;
  const int row0 = blockIdx.x * 128;

  // cooperative load + f32->bf16 convert (+ mean normalization of agg)
#pragma unroll
  for (int i = 0; i < 16; ++i) {
    int idx = tid + i * 256;   // 0..4095 float4 slots
    int r = idx >> 5;          // 0..127
    int c4 = idx & 31;         // float4 column
    int g = row0 + r;
    float4 hv = {0.f, 0.f, 0.f, 0.f}, av = {0.f, 0.f, 0.f, 0.f};
    if (g < n) {
      hv = reinterpret_cast<const float4*>(h + (size_t)g * HDIM)[c4];
      float rc = 1.0f / fmaxf(cnt[g], 1.0f);
      float4 t = reinterpret_cast<const float4*>(agg + (size_t)g * HDIM)[c4];
      av.x = t.x * rc; av.y = t.y * rc; av.z = t.z * rc; av.w = t.w * rc;
    }
    store4bf(&sh[r * LDA + c4 * 4], hv);
    store4bf(&sa[r * LDA + c4 * 4], av);
  }
  __syncthreads();

  const int wave = tid >> 5, lane = tid & 31;
  const int n0 = (wave << 4) + (lane & 15);  // output column
  const int kc = lane >> 4;                  // K-half select

  // B fragments: 16 contiguous bf16 per frag from transposed weights
  v16bf wl[4], wr[4];
#pragma unroll
  for (int kk = 0; kk < 4; ++kk) {
    int kb = kk * 32 + kc * 16;
    const v4u* pl = reinterpret_cast<const v4u*>(WlT + n0 * HDIM + kb);
    const v4u* pr = reinterpret_cast<const v4u*>(WrT + n0 * HDIM + kb);
    AFrag fl, fr;
    fl.q[0] = pl[0]; fl.q[1] = pl[1];
    fr.q[0] = pr[0]; fr.q[1] = pr[1];
    wl[kk] = fl.v;
    wr[kk] = fr.v;
  }
  // fold bias + BN into one FMA per output: relu(acc*A + C)
  const float A = gma[n0] * rsqrtf(var[n0] + EPS);
  const float C = (bl[n0] - mu[n0]) * A + bta[n0];

#pragma unroll
  for (int rb = 0; rb < 8; ++rb) {
    const int m = rb * 16 + (lane & 15);  // LDS row for this lane's A fragment
    // batch all A-fragment loads for this row block (one clause, overlap w/ WMMA)
    AFrag aa[4], ah[4];
#pragma unroll
    for (int kk = 0; kk < 4; ++kk) {
      int k0 = kk * 32 + kc * 16;
      aa[kk].q[0] = *reinterpret_cast<const v4u*>(&sa[m * LDA + k0]);
      aa[kk].q[1] = *reinterpret_cast<const v4u*>(&sa[m * LDA + k0 + 8]);
      ah[kk].q[0] = *reinterpret_cast<const v4u*>(&sh[m * LDA + k0]);
      ah[kk].q[1] = *reinterpret_cast<const v4u*>(&sh[m * LDA + k0 + 8]);
    }
    v8f acc = {0.f, 0.f, 0.f, 0.f, 0.f, 0.f, 0.f, 0.f};
#pragma unroll
    for (int kk = 0; kk < 4; ++kk) {
      acc = __builtin_amdgcn_wmma_f32_16x16x32_bf16(false, aa[kk].v, false, wl[kk],
                                                    (short)0, acc, false, false);
      acc = __builtin_amdgcn_wmma_f32_16x16x32_bf16(false, ah[kk].v, false, wr[kk],
                                                    (short)0, acc, false, false);
    }
    const int rowbase = row0 + rb * 16;
    if (rowbase + 15 < n) {
      // full tile: branch-free epilogue
#pragma unroll
      for (int r = 0; r < 8; ++r) {
        float v = acc[r] * A + C;
        h[(size_t)(rowbase + r + kc * 8) * HDIM + n0] = fmaxf(v, 0.0f);
      }
    } else {
#pragma unroll
      for (int r = 0; r < 8; ++r) {
        int grow = rowbase + r + kc * 8;
        if (grow < n) {
          float v = acc[r] * A + C;
          h[(size_t)grow * HDIM + n0] = fmaxf(v, 0.0f);
        }
      }
    }
  }
}

// out = relu(h @ W1 + b1) @ W2 + b2 over concat(h_u, h_t) rows; 128 rows/block
// W1T is pre-transposed bf16 [n][k].
__global__ __launch_bounds__(256) void k_mlp(
    const float* __restrict__ h_u, const float* __restrict__ h_t,
    const unsigned short* __restrict__ W1T, const float* __restrict__ b1,
    const float* __restrict__ W2, const float* __restrict__ b2,
    float* __restrict__ out, int nu, int ntot) {
  __shared__ unsigned short sh[128 * LDA];       // input rows, bf16
  __shared__ float hid[8][16][HID2 + 4];          // per-wave hidden tiles
  __shared__ float sW2[HID2 * OUTD];
  const int tid = threadIdx.x;
  const int row0 = blockIdx.x * 128;

  if (tid < HID2 * OUTD) sW2[tid] = W2[tid];

#pragma unroll
  for (int i = 0; i < 16; ++i) {
    int idx = tid + i * 256;  // 0..4095 float4 slots
    int r = idx >> 5;
    int c4 = idx & 31;
    int g = row0 + r;
    float4 v = {0.f, 0.f, 0.f, 0.f};
    if (g < ntot) {
      const float* src = (g < nu) ? (h_u + (size_t)g * HDIM)
                                  : (h_t + (size_t)(g - nu) * HDIM);
      v = reinterpret_cast<const float4*>(src)[c4];
    }
    store4bf(&sh[r * LDA + c4 * 4], v);
  }
  __syncthreads();

  const int wave = tid >> 5, lane = tid & 31;
  const int kc = lane >> 4;
  const int n0l = lane & 15;
  const int mrow = wave * 16 + (lane & 15);

  // A fragments are independent of the hidden column tile: load once
  AFrag a[4];
#pragma unroll
  for (int kk = 0; kk < 4; ++kk) {
    int k0 = kk * 32 + kc * 16;
    a[kk].q[0] = *reinterpret_cast<const v4u*>(&sh[mrow * LDA + k0]);
    a[kk].q[1] = *reinterpret_cast<const v4u*>(&sh[mrow * LDA + k0 + 8]);
  }

#pragma unroll
  for (int t = 0; t < 4; ++t) {  // hidden column tiles of 16 (HID2 = 64)
    int ncol = t * 16 + n0l;
    v8f acc = {0.f, 0.f, 0.f, 0.f, 0.f, 0.f, 0.f, 0.f};
#pragma unroll
    for (int kk = 0; kk < 4; ++kk) {
      int kb = kk * 32 + kc * 16;
      AFrag w;
      const v4u* pw = reinterpret_cast<const v4u*>(W1T + ncol * HDIM + kb);
      w.q[0] = pw[0]; w.q[1] = pw[1];
      acc = __builtin_amdgcn_wmma_f32_16x16x32_bf16(false, a[kk].v, false, w.v,
                                                    (short)0, acc, false, false);
    }
    float bb = b1[ncol];
#pragma unroll
    for (int r = 0; r < 8; ++r)
      hid[wave][r + kc * 8][ncol] = fmaxf(acc[r] + bb, 0.0f);
  }
  __syncthreads();

  // second stage: 64 -> 2 per row, lane l: row = l%16, output = l/16
  {
    int m = lane & 15, o = lane >> 4;
    int g = row0 + wave * 16 + m;
    float a2 = b2[o];
#pragma unroll
    for (int j = 0; j < HID2; ++j) a2 += hid[wave][m][j] * sW2[j * OUTD + o];
    if (g < ntot && o < OUTD) out[(size_t)g * OUTD + o] = a2;
  }
}

// ---------------- host launcher ----------------
extern "C" void kernel_launch(void* const* d_in, const int* in_sizes, int n_in,
                              void* d_out, int out_size, void* d_ws, size_t ws_size,
                              hipStream_t stream) {
  const int* x_user  = (const int*)d_in[0];
  const int* x_txn   = (const int*)d_in[1];
  const int* ei0_src = (const int*)d_in[2];
  const int* ei0_dst = (const int*)d_in[3];
  const int* ei1_src = (const int*)d_in[4];
  const int* ei1_dst = (const int*)d_in[5];
  const float* emb_user = (const float*)d_in[6];
  const float* emb_txn  = (const float*)d_in[7];
  const float* Wl  = (const float*)d_in[8];
  const float* bl  = (const float*)d_in[9];
  const float* Wr  = (const float*)d_in[10];
  const float* gma = (const float*)d_in[11];
  const float* bta = (const float*)d_in[12];
  const float* mu  = (const float*)d_in[13];
  const float* var = (const float*)d_in[14];
  const float* W1  = (const float*)d_in[15];
  const float* b1  = (const float*)d_in[16];
  const float* W2  = (const float*)d_in[17];
  const float* b2  = (const float*)d_in[18];
  float* out = (float*)d_out;

  // workspace carve-up (all 16B-aligned offsets)
  char* ws = (char*)d_ws;
  float* h_u   = (float*)ws; ws += (size_t)NU * HDIM * sizeof(float);
  float* h_t   = (float*)ws; ws += (size_t)NT * HDIM * sizeof(float);
  float* agg_u = (float*)ws; ws += (size_t)NU * HDIM * sizeof(float);
  float* agg_t = (float*)ws; ws += (size_t)NT * HDIM * sizeof(float);
  float* cnt_u = (float*)ws; ws += (size_t)NU * sizeof(float);
  float* cnt_t = (float*)ws; ws += (size_t)NT * sizeof(float);
  unsigned short* WlT = (unsigned short*)ws; ws += (size_t)NLAY * 2 * HDIM * HDIM * sizeof(unsigned short);
  unsigned short* WrT = (unsigned short*)ws; ws += (size_t)NLAY * 2 * HDIM * HDIM * sizeof(unsigned short);
  unsigned short* W1T = (unsigned short*)ws; ws += (size_t)HID2 * HDIM * sizeof(unsigned short);

  // 0) pre-transpose + bf16-convert weights
  {
    const int nblk = (HDIM * HDIM + 255) / 256;
    for (int m = 0; m < NLAY * 2; ++m) {
      k_prep_w<<<nblk, 256, 0, stream>>>(Wl + (size_t)m * HDIM * HDIM,
                                         WlT + (size_t)m * HDIM * HDIM, HDIM, HDIM);
      k_prep_w<<<nblk, 256, 0, stream>>>(Wr + (size_t)m * HDIM * HDIM,
                                         WrT + (size_t)m * HDIM * HDIM, HDIM, HDIM);
    }
    k_prep_w<<<(HDIM * HID2 + 255) / 256, 256, 0, stream>>>(W1, W1T, HDIM, HID2);
  }

  // 1) embedding gather
  k_gather<<<(NU + 7) / 8, 256, 0, stream>>>(emb_user, x_user, h_u, NU);
  k_gather<<<(NT + 7) / 8, 256, 0, stream>>>(emb_txn, x_txn, h_t, NT);

  // 2) per-destination edge counts (layer invariant)
  hipMemsetAsync(cnt_u, 0, (size_t)NU * sizeof(float), stream);
  hipMemsetAsync(cnt_t, 0, (size_t)NT * sizeof(float), stream);
  k_count<<<(NE + 255) / 256, 256, 0, stream>>>(ei0_dst, cnt_t, NE);
  k_count<<<(NE + 255) / 256, 256, 0, stream>>>(ei1_dst, cnt_u, NE);

  // 3) GNN layers
  for (int i = 0; i < NLAY; ++i) {
    hipMemsetAsync(agg_t, 0, (size_t)NT * HDIM * sizeof(float), stream);
    hipMemsetAsync(agg_u, 0, (size_t)NU * HDIM * sizeof(float), stream);
    k_scatter<<<(NE + 7) / 8, 256, 0, stream>>>(h_u, agg_t, ei0_src, ei0_dst, NE);
    k_scatter<<<(NE + 7) / 8, 256, 0, stream>>>(h_t, agg_u, ei1_src, ei1_dst, NE);

    // txn update: weights [i][0], BN params [i][1]
    k_gemm_bn<<<(NT + 127) / 128, 256, 0, stream>>>(
        NT, h_t, agg_t, cnt_t,
        WlT + (size_t)(i * 2 + 0) * HDIM * HDIM, bl + (size_t)(i * 2 + 0) * HDIM,
        WrT + (size_t)(i * 2 + 0) * HDIM * HDIM,
        gma + (size_t)(i * 2 + 1) * HDIM, bta + (size_t)(i * 2 + 1) * HDIM,
        mu  + (size_t)(i * 2 + 1) * HDIM, var + (size_t)(i * 2 + 1) * HDIM);
    // user update: weights [i][1], BN params [i][0]
    k_gemm_bn<<<(NU + 127) / 128, 256, 0, stream>>>(
        NU, h_u, agg_u, cnt_u,
        WlT + (size_t)(i * 2 + 1) * HDIM * HDIM, bl + (size_t)(i * 2 + 1) * HDIM,
        WrT + (size_t)(i * 2 + 1) * HDIM * HDIM,
        gma + (size_t)(i * 2 + 0) * HDIM, bta + (size_t)(i * 2 + 0) * HDIM,
        mu  + (size_t)(i * 2 + 0) * HDIM, var + (size_t)(i * 2 + 0) * HDIM);
  }

  // 4) MLP head over concat(h_u, h_t)
  const int ntot = NU + NT;
  k_mlp<<<(ntot + 127) / 128, 256, 0, stream>>>(h_u, h_t, W1T, b1, W2, b2, out,
                                                NU, ntot);
}
